// Decoder_77206332113287
// MI455X (gfx1250) — matrix operand order repaired and checked by reading, not compiled, not run on hardware
//
#include <hip/hip_runtime.h>
#include <hip/hip_bf16.h>
#include <math.h>

// ---------------------------------------------------------------------------
// Decoder: GRU (1 layer) + masked attention + output projection + log_softmax
// All GEMMs use v_wmma_f32_16x16x32_f16 (CDNA5 / gfx1250, wave32).
// Round 2: 4x4 register blocking (64x64 per wave) in the dominant output
// projection GEMM to cut L2 fragment traffic 4x and raise WMMA density.
// ---------------------------------------------------------------------------

typedef __attribute__((ext_vector_type(16))) _Float16 v16h;
typedef __attribute__((ext_vector_type(8)))  float    v8f;

#define NB   16      // batch
#define TDEC 64
#define TENC 512
#define EDIM 512
#define HDIM 512
#define VOC  32000
#define G3   1536    // 3*H
#define OCW  1024    // oc row width = 2*H

__device__ __forceinline__ int lane_id() { return (int)(threadIdx.x & 31u); }
__device__ __forceinline__ int wave_id() { return (int)(threadIdx.x >> 5); }

__device__ __forceinline__ v8f wmma_f16(v16h a, v16h b, v8f c) {
  // (neg_a, A, neg_b, B, c_mod, C, reuse_a, reuse_b)
  return __builtin_amdgcn_wmma_f32_16x16x32_f16(false, a, false, b, (short)0, c,
                                                false, false);
}

// A fragment: 16x32 (MxK), f32 source row-major, leading dim lda.
// lanes 0-15 -> K 0..15 ; lanes 16-31 -> K 16..31 (contiguous, matches B loaders).
__device__ __forceinline__ v16h load_A(const float* base, int lda) {
  int l = lane_id();
  int m = l & 15;
  int hi = l >> 4;
  const float* p = base + m * lda;
  v16h a;
#pragma unroll
  for (int i = 0; i < 8; ++i) a[i] = (_Float16)p[hi * 16 + i];
#pragma unroll
  for (int i = 0; i < 8; ++i) a[8 + i] = (_Float16)p[hi * 16 + 8 + i];
  return a;
}

// B fragment: 32x16 (KxN) where B[k][n] = W[n*ldw + k]  (i.e. computing X @ W^T)
// lanes 0-15 hold K 0..15, lanes 16-31 hold K 16..31; N = lane&15. Contiguous runs.
__device__ __forceinline__ v16h load_B_T(const float* W, int ldw) {
  int l = lane_id();
  int n = l & 15;
  int hi = l >> 4;
  const float* p = W + n * ldw + hi * 16;
  v16h b;
#pragma unroll
  for (int i = 0; i < 16; ++i) b[i] = (_Float16)p[i];
  return b;
}

// B fragment: 32x16 (KxN) where B[k][n] = S[k*lds + n]  (row-major KxN source)
__device__ __forceinline__ v16h load_B_N(const float* S, int lds) {
  int l = lane_id();
  int n = l & 15;
  int hi = l >> 4;
  const float* p = S + (hi * 16) * lds + n;
  v16h b;
#pragma unroll
  for (int i = 0; i < 16; ++i) b[i] = (_Float16)p[i * lds];
  return b;
}

// ---------------------------------------------------------------------------
// Kernel 1: xg = emb @ W_ih^T + b_ih      [1024 x 1536], K = 512
// ---------------------------------------------------------------------------
__global__ __launch_bounds__(256) void k_xg(const float* __restrict__ emb,
                                            const float* __restrict__ W_ih,
                                            const float* __restrict__ b_ih,
                                            float* __restrict__ xg) {
  int tile = blockIdx.x * 8 + wave_id();      // 6144 tiles total
  int mt = tile / (G3 / 16);
  int nt = tile % (G3 / 16);
  const float* Abase = emb + (mt * 16) * EDIM;
  const float* Bbase = W_ih + (nt * 16) * EDIM;
  v8f acc = {};
  for (int k = 0; k < EDIM; k += 32) {
    v16h a = load_A(Abase + k, EDIM);
    v16h b = load_B_T(Bbase + k, EDIM);
    acc = wmma_f16(a, b, acc);
  }
  int l = lane_id(), n = l & 15, mo = (l >> 4) * 8;
  float bias = b_ih[nt * 16 + n];
  float* D = xg + (mt * 16) * G3 + nt * 16 + n;
#pragma unroll
  for (int v = 0; v < 8; ++v) D[(mo + v) * G3] = acc[v] + bias;
}

// ---------------------------------------------------------------------------
// Kernel 2: GRU recurrence. One workgroup (32 waves), 64 sequential steps.
// h [16x512] and gh [16x1536] live in LDS (128 KB of the 320 KB WGP pool).
// Writes GRU outputs to oc[:, 0:512], final h to h_out.
// ---------------------------------------------------------------------------
__global__ __launch_bounds__(1024) void k_gru(const float* __restrict__ h0,
                                              const float* __restrict__ xg,
                                              const float* __restrict__ W_hh,
                                              const float* __restrict__ b_hh,
                                              float* __restrict__ oc,
                                              float* __restrict__ h_out) {
  extern __shared__ float smem[];
  float* hbuf  = smem;                 // 16*512 f32
  float* ghbuf = smem + NB * HDIM;     // 16*1536 f32
  int tid = (int)threadIdx.x;
  for (int i = tid; i < NB * HDIM; i += 1024) hbuf[i] = h0[i];
  __syncthreads();
  int w = wave_id();
  for (int t = 0; t < TDEC; ++t) {
    // gh = h @ W_hh^T : 96 N-tiles, 3 per wave, K = 512
#pragma unroll
    for (int q = 0; q < 3; ++q) {
      int nt = w * 3 + q;
      v8f acc = {};
      for (int k = 0; k < HDIM; k += 32) {
        v16h a = load_A(hbuf + k, HDIM);
        v16h b = load_B_T(W_hh + (nt * 16) * HDIM + k, HDIM);
        acc = wmma_f16(a, b, acc);
      }
      int l = lane_id(), n = l & 15, mo = (l >> 4) * 8;
      float* D = ghbuf + nt * 16 + n;
#pragma unroll
      for (int v = 0; v < 8; ++v) D[(mo + v) * G3] = acc[v];
    }
    __syncthreads();
    // fused gates (PyTorch convention: r,z,n)
    for (int i = tid; i < NB * HDIM; i += 1024) {
      int b = i >> 9, j = i & (HDIM - 1);
      const float* xrow = xg + (size_t)(b * TDEC + t) * G3;
      float xr = xrow[j], xz = xrow[HDIM + j], xn = xrow[2 * HDIM + j];
      float hr = ghbuf[b * G3 + j]            + b_hh[j];
      float hz = ghbuf[b * G3 + HDIM + j]     + b_hh[HDIM + j];
      float hn = ghbuf[b * G3 + 2 * HDIM + j] + b_hh[2 * HDIM + j];
      float r = 1.f / (1.f + __expf(-(xr + hr)));
      float z = 1.f / (1.f + __expf(-(xz + hz)));
      float nn = tanhf(xn + r * hn);
      float hnew = (1.f - z) * nn + z * hbuf[i];
      hbuf[i] = hnew;
      oc[(size_t)(b * TDEC + t) * OCW + j] = hnew;
    }
    __syncthreads();
  }
  for (int i = tid; i < NB * HDIM; i += 1024) h_out[i] = hbuf[i];
}

// ---------------------------------------------------------------------------
// Kernel 3a: scores s[b,j,t] = hs[b,j,:] . out[b,t,:], masked to -inf
// Per b: M=512(j), N=64(t), K=512. Store to ws scores [b][j][t].
// ---------------------------------------------------------------------------
__global__ __launch_bounds__(256) void k_scores(const float* __restrict__ hs,
                                                const float* __restrict__ oc,
                                                const int* __restrict__ mask,
                                                float* __restrict__ sc) {
  int tile = blockIdx.x * 8 + wave_id();      // 2048 tiles
  int b = tile / 128;
  int r = tile % 128;
  int mt = r / 4, nt = r % 4;
  const float* Abase = hs + (size_t)(b * TENC + mt * 16) * HDIM;
  const float* Bbase = oc + (size_t)(b * TDEC + nt * 16) * OCW; // K contiguous rows
  v8f acc = {};
  for (int k = 0; k < HDIM; k += 32) {
    v16h a  = load_A(Abase + k, HDIM);
    v16h bb = load_B_T(Bbase + k, OCW);
    acc = wmma_f16(a, bb, acc);
  }
  int l = lane_id(), n = l & 15, mo = (l >> 4) * 8;
  int mlim = mask[b];
  float* D = sc + (size_t)(b * TENC + mt * 16) * TDEC + nt * 16 + n;
#pragma unroll
  for (int v = 0; v < 8; ++v) {
    int j = mt * 16 + mo + v;
    D[(mo + v) * TDEC] = (j < mlim) ? acc[v] : -__builtin_inff();
  }
}

// ---------------------------------------------------------------------------
// Kernel 3b: softmax over encoder dim j for each (b,t) column.
// Writes attn [b,j,t] to d_out and attnT [b,t,j] to ws for the context GEMM.
// ---------------------------------------------------------------------------
__global__ __launch_bounds__(256) void k_softmax(const float* __restrict__ sc,
                                                 float* __restrict__ attn,
                                                 float* __restrict__ attnT) {
  int bt = (int)blockIdx.x;                  // b*64 + t
  int b = bt >> 6, t = bt & 63;
  const float* col = sc + (size_t)(b * TENC) * TDEC + t;   // stride TDEC
  __shared__ float red[256];
  int tid = (int)threadIdx.x;
  float v0 = col[tid * TDEC];
  float v1 = col[(tid + 256) * TDEC];
  red[tid] = fmaxf(v0, v1);
  __syncthreads();
  for (int s = 128; s > 0; s >>= 1) {
    if (tid < s) red[tid] = fmaxf(red[tid], red[tid + s]);
    __syncthreads();
  }
  float mx = red[0];
  __syncthreads();
  float e0 = __expf(v0 - mx), e1 = __expf(v1 - mx);   // exp(-inf)=0 for masked
  red[tid] = e0 + e1;
  __syncthreads();
  for (int s = 128; s > 0; s >>= 1) {
    if (tid < s) red[tid] += red[tid + s];
    __syncthreads();
  }
  float inv = 1.f / red[0];
  float a0 = e0 * inv, a1 = e1 * inv;
  attn[(size_t)(b * TENC + tid) * TDEC + t]        = a0;
  attn[(size_t)(b * TENC + tid + 256) * TDEC + t]  = a1;
  attnT[(size_t)(b * TDEC + t) * TENC + tid]       = a0;
  attnT[(size_t)(b * TDEC + t) * TENC + tid + 256] = a1;
}

// ---------------------------------------------------------------------------
// Kernel 3c: context c[b,t,h] = sum_j attnT[b,t,j] * hs[b,j,h]
// Per b: M=64(t), N=512(h), K=512(j). Writes into oc[:, 512:1024].
// ---------------------------------------------------------------------------
__global__ __launch_bounds__(256) void k_ctx(const float* __restrict__ attnT,
                                             const float* __restrict__ hs,
                                             float* __restrict__ oc) {
  int tile = blockIdx.x * 8 + wave_id();      // 2048 tiles
  int b = tile / 128;
  int r = tile % 128;
  int mt = r / 32, nt = r % 32;
  const float* Abase = attnT + (size_t)(b * TDEC + mt * 16) * TENC;
  const float* Bbase = hs + (size_t)(b * TENC) * HDIM + nt * 16;
  v8f acc = {};
  for (int k = 0; k < TENC; k += 32) {
    v16h a  = load_A(Abase + k, TENC);
    v16h bb = load_B_N(Bbase + (size_t)k * HDIM, HDIM);
    acc = wmma_f16(a, bb, acc);
  }
  int l = lane_id(), n = l & 15, mo = (l >> 4) * 8;
  float* D = oc + (size_t)(b * TDEC + mt * 16) * OCW + HDIM + nt * 16 + n;
#pragma unroll
  for (int v = 0; v < 8; ++v) D[(mo + v) * OCW] = acc[v];
}

// ---------------------------------------------------------------------------
// Kernel 4a: logits = oc @ W_out^T + b_out   [1024 x 32000], K = 1024
// Dominant GEMM (67 GF, 131 MB weights). 4x4 register blocking: each wave
// computes a 64x64 output tile (16 f32 accumulators = 128 VGPRs); every A/B
// fragment feeds 4 WMMAs, cutting L2 fragment traffic 4x vs 1x1 tiling.
// ---------------------------------------------------------------------------
__global__ __launch_bounds__(256) void k_logits(const float* __restrict__ oc,
                                                const float* __restrict__ W_out,
                                                const float* __restrict__ b_out,
                                                float* __restrict__ logits) {
  int wt = blockIdx.x * 8 + wave_id();        // 8000 wave tiles (16 x 500)
  int mb = wt / (VOC / 64);                   // 0..15  (64 rows each)
  int nb = wt % (VOC / 64);                   // 0..499 (64 cols each)
  const float* Abase = oc + (size_t)(mb * 64) * OCW;
  const float* Bbase = W_out + (size_t)(nb * 64) * OCW;
  v8f acc[4][4] = {};
  for (int k = 0; k < OCW; k += 32) {
    v16h a0 = load_A(Abase + 0 * 16 * OCW + k, OCW);
    v16h a1 = load_A(Abase + 1 * 16 * OCW + k, OCW);
    v16h a2 = load_A(Abase + 2 * 16 * OCW + k, OCW);
    v16h a3 = load_A(Abase + 3 * 16 * OCW + k, OCW);
#pragma unroll
    for (int q = 0; q < 4; ++q) {
      v16h bb = load_B_T(Bbase + (size_t)(q * 16) * OCW + k, OCW);
      acc[0][q] = wmma_f16(a0, bb, acc[0][q]);
      acc[1][q] = wmma_f16(a1, bb, acc[1][q]);
      acc[2][q] = wmma_f16(a2, bb, acc[2][q]);
      acc[3][q] = wmma_f16(a3, bb, acc[3][q]);
    }
  }
  int l = lane_id(), n = l & 15, mo = (l >> 4) * 8;
#pragma unroll
  for (int q = 0; q < 4; ++q) {
    float bias = b_out[nb * 64 + q * 16 + n];
#pragma unroll
    for (int r = 0; r < 4; ++r) {
      float* D = logits + (size_t)(mb * 64 + r * 16) * VOC + nb * 64 + q * 16 + n;
#pragma unroll
      for (int v = 0; v < 8; ++v) D[(size_t)(mo + v) * VOC] = acc[r][q][v] + bias;
    }
  }
}

// ---------------------------------------------------------------------------
// Kernel 4b: in-place log_softmax over V per row (1024 rows x 32000).
// ---------------------------------------------------------------------------
__global__ __launch_bounds__(256) void k_logsoftmax(float* __restrict__ logp) {
  int row = (int)blockIdx.x;
  float* p = logp + (size_t)row * VOC;
  __shared__ float red[256];
  int tid = (int)threadIdx.x;
  float mx = -3.4e38f;
  for (int i = tid; i < VOC; i += 256) mx = fmaxf(mx, p[i]);
  red[tid] = mx;
  __syncthreads();
  for (int s = 128; s > 0; s >>= 1) {
    if (tid < s) red[tid] = fmaxf(red[tid], red[tid + s]);
    __syncthreads();
  }
  mx = red[0];
  __syncthreads();
  float sum = 0.f;
  for (int i = tid; i < VOC; i += 256) sum += __expf(p[i] - mx);
  red[tid] = sum;
  __syncthreads();
  for (int s = 128; s > 0; s >>= 1) {
    if (tid < s) red[tid] += red[tid + s];
    __syncthreads();
  }
  float lse = mx + __logf(red[0]);
  for (int i = tid; i < VOC; i += 256) p[i] = p[i] - lse;
}

// ---------------------------------------------------------------------------
extern "C" void kernel_launch(void* const* d_in, const int* in_sizes, int n_in,
                              void* d_out, int out_size, void* d_ws, size_t ws_size,
                              hipStream_t stream) {
  (void)in_sizes; (void)n_in; (void)out_size; (void)ws_size;
  const float* emb   = (const float*)d_in[0];   // [16,64,512]
  const float* hs    = (const float*)d_in[1];   // [16,512,512]
  const float* h0    = (const float*)d_in[2];   // [1,16,512]
  const int*   mask  = (const int*)d_in[3];     // [16]
  const float* W_ih  = (const float*)d_in[4];   // [1536,512]
  const float* W_hh  = (const float*)d_in[5];   // [1536,512]
  const float* b_ih  = (const float*)d_in[6];   // [1536]
  const float* b_hh  = (const float*)d_in[7];   // [1536]
  const float* W_out = (const float*)d_in[8];   // [32000,1024]
  const float* b_out = (const float*)d_in[9];   // [32000]

  // Output layout: logp [16,64,32000] | h_new [1,16,512] | attn [16,512,64]
  float* out   = (float*)d_out;
  float* logp  = out;
  float* h_new = out + (size_t)NB * TDEC * VOC;
  float* attn  = h_new + (size_t)NB * HDIM;

  // Workspace layout (~14.7 MB)
  float* xg    = (float*)d_ws;                    // 1024 x 1536
  float* oc    = xg + (size_t)NB * TDEC * G3;     // 1024 x 1024 (gru out | ctx)
  float* sc    = oc + (size_t)NB * TDEC * OCW;    // 16 x 512 x 64
  float* attnT = sc + (size_t)NB * TENC * TDEC;   // 16 x 64 x 512

  // Allow 128 KB dynamic LDS for the persistent GRU workgroup (not a stream op,
  // safe under graph capture; idempotent/deterministic).
  hipFuncSetAttribute((const void*)k_gru,
                      hipFuncAttributeMaxDynamicSharedMemorySize,
                      (NB * HDIM + NB * G3) * (int)sizeof(float));

  k_xg<<<768, 256, 0, stream>>>(emb, W_ih, b_ih, xg);
  k_gru<<<1, 1024, (NB * HDIM + NB * G3) * sizeof(float), stream>>>(
      h0, xg, W_hh, b_hh, oc, h_new);
  k_scores<<<256, 256, 0, stream>>>(hs, oc, mask, sc);
  k_softmax<<<NB * TDEC, 256, 0, stream>>>(sc, attn, attnT);
  k_ctx<<<256, 256, 0, stream>>>(attnT, hs, oc);
  k_logits<<<1000, 256, 0, stream>>>(oc, W_out, b_out, logp);
  k_logsoftmax<<<NB * TDEC, 256, 0, stream>>>(logp);
}
